// hhgnnConv_19834158973283
// MI455X (gfx1250) — matrix operand order repaired and checked by reading, not compiled, not run on hardware
//
#include <hip/hip_runtime.h>
#include <hip/hip_bf16.h>
#include <math.h>
#include <stdint.h>

// Problem constants (match reference)
#define N_NODES 100000
#define E_EDGES 50000
#define NNZ_E   1000000
#define IN_CH   128
#define HEADS   8
#define CH      16
#define HC      128   // HEADS*CH
#define NEG_SLOPE 0.2f

typedef __attribute__((ext_vector_type(16))) _Float16 v16h;
typedef __attribute__((ext_vector_type(8)))  float    v8f;

// ---------------------------------------------------------------------------
// utility: fill buffer with constant
// ---------------------------------------------------------------------------
__global__ void fill_kernel(float* __restrict__ p, float v, int n) {
    int i = blockIdx.x * blockDim.x + threadIdx.x;
    if (i < n) p[i] = v;
}

// ---------------------------------------------------------------------------
// float atomic max via int/uint punning (monotone over IEEE-754 bit order).
// emax/vmax are initialized to -inf, so both branches compose correctly.
// ---------------------------------------------------------------------------
__device__ inline void atomicMaxF32(float* addr, float val) {
    if (val >= 0.0f)
        atomicMax((int*)addr, __float_as_int(val));
    else
        atomicMin((unsigned int*)addr, (unsigned int)__float_as_int(val));
}

// ---------------------------------------------------------------------------
// Pack Ww (128x128 f32, row-major [k][n]) into f16 B-fragment order:
// Wh[((ct*4 + s)*32 + lane)*16 + t] = f16( Ww[s*32 + 16*(lane>>4) + t][ct*16 + (lane&15)] )
// so each wave's per-k-step B fragment is one contiguous 32-byte load.
// ---------------------------------------------------------------------------
__global__ void pack_w_kernel(const float* __restrict__ Ww, _Float16* __restrict__ Wh) {
    int idx = blockIdx.x * blockDim.x + threadIdx.x;
    if (idx >= HC * IN_CH) return;
    const int t    = idx & 15;
    const int lane = (idx >> 4) & 31;
    const int s    = (idx >> 9) & 3;
    const int ct   = idx >> 11;
    const int hi   = lane >> 4;
    const int n    = ct * 16 + (lane & 15);
    const int k    = s * 32 + 16 * hi + t;
    Wh[idx] = (_Float16)Ww[(size_t)k * HC + n];
}

// ---------------------------------------------------------------------------
// GEMM: Xn = X @ Ww + Wb   (M=100000, K=128, N=128)
// Block = 256 threads = 8 waves; block stages one 16x128 f32 X tile into LDS
// via global_load_async_to_lds_b128 (ASYNCcnt), then each wave computes one
// 16x16 output tile with 4 chained v_wmma_f32_16x16x32_f16.
// ---------------------------------------------------------------------------
__global__ __launch_bounds__(256) void gemm_wmma_kernel(
    const float* __restrict__ X, const _Float16* __restrict__ Wh,
    const float* __restrict__ Wb, float* __restrict__ Xn) {

    // 132-float row pitch: A-fragment reads spread over all 64 LDS banks
    __shared__ float tileX[16][132];

    const int tid  = threadIdx.x;
    const int lane = tid & 31;
    const int wave = tid >> 5;
    const int m    = lane & 15;   // row within tile (A) / col within tile (B,C)
    const int hi   = lane >> 4;   // half-wave select
    const int row0 = blockIdx.x * 16;
    const int col0 = wave * 16;

    // ---- async-copy X tile (16 rows x 512B) into LDS: 512 x 16B chunks ----
    const uint32_t ldsBase = (uint32_t)(uintptr_t)(&tileX[0][0]);
#pragma unroll
    for (int it = 0; it < 2; ++it) {
        const int q   = tid + 256 * it;   // 16B-chunk id, 0..511
        const int r   = q >> 5;           // tile row (32 chunks per row)
        const int c16 = q & 31;           // chunk within row
        const uint64_t gaddr =
            (uint64_t)(uintptr_t)(X + (size_t)(row0 + r) * IN_CH + c16 * 4);
        const uint32_t loff = ldsBase + r * (132 * 4) + c16 * 16;
        asm volatile("global_load_async_to_lds_b128 %0, %1, off"
                     :: "v"(loff), "v"(gaddr) : "memory");
    }
    asm volatile("s_wait_asynccnt 0" ::: "memory");
    __syncthreads();

    v8f acc = {};
#pragma unroll
    for (int s = 0; s < 4; ++s) {
        const int k0 = 32 * s;
        v16h a;
        // A: 16x32 f16 tile. Lane (m,hi): halves 0..15 = K[k0+16*hi .. +15]
        //    (contiguous order, matching packed B)
#pragma unroll
        for (int t = 0; t < 8; ++t) {
            a[t]     = (_Float16)tileX[m][k0 + 16 * hi + t];
            a[8 + t] = (_Float16)tileX[m][k0 + 16 * hi + 8 + t];
        }
        // B fragment: one contiguous 32-byte pre-packed f16 load
        const v16h b = *(const v16h*)(Wh + (((size_t)wave * 4 + s) * 32 + lane) * 16);

        acc = __builtin_amdgcn_wmma_f32_16x16x32_f16(
            /*neg_a=*/false, a, /*neg_b=*/false, b,
            /*c_mod=*/(short)0, acc, /*reuse_a=*/false, /*reuse_b=*/false);
    }

    const int   col  = col0 + m;
    const float bias = Wb[col];
#pragma unroll
    for (int r = 0; r < 8; ++r) {
        const int row = row0 + r + 8 * hi;   // C/D layout: VGPR r -> M=r / r+8
        Xn[(size_t)row * HC + col] = acc[r] + bias;
    }
}

// ---------------------------------------------------------------------------
// Attention logits: one thread per (nnz, head).
// val = leaky_relu( dot16( F[gidx[nz]][h][:], att[cls[nz]][h][:] ) )
// logit[i] = val ; segment-max into smax[sidx[nz]*H + h]
// ---------------------------------------------------------------------------
__global__ void seg_logits_kernel(
    const float* __restrict__ F, const float* __restrict__ att,
    const int* __restrict__ gidx, const int* __restrict__ sidx,
    const int* __restrict__ cls,
    float* __restrict__ logit, float* __restrict__ smax) {

    int i = blockIdx.x * blockDim.x + threadIdx.x;
    if (i >= NNZ_E * HEADS) return;
    const int nz = i >> 3, h = i & 7;
    const int g  = gidx[nz];
    const int k  = cls[nz];

    const float4* xr = (const float4*)(F + (size_t)g * HC + h * CH);
    const float4* ar = (const float4*)(att + ((size_t)k * HEADS + h) * CH);
    float s = 0.0f;
#pragma unroll
    for (int q = 0; q < 4; ++q) {
        float4 x = xr[q], a = ar[q];
        s += x.x * a.x + x.y * a.y + x.z * a.z + x.w * a.w;
    }
    const float val = (s >= 0.0f) ? s : NEG_SLOPE * s;
    logit[i] = val;
    atomicMaxF32(&smax[(size_t)sidx[nz] * HEADS + h], val);
}

// ---------------------------------------------------------------------------
// ex = exp(logit - smax[seg]); logit <- ex; segment-sum into ssum
// ---------------------------------------------------------------------------
__global__ void seg_exp_kernel(
    const int* __restrict__ sidx, const float* __restrict__ smax,
    float* __restrict__ logit, float* __restrict__ ssum) {

    int i = blockIdx.x * blockDim.x + threadIdx.x;
    if (i >= NNZ_E * HEADS) return;
    const int nz = i >> 3, h = i & 7;
    const size_t s = (size_t)sidx[nz] * HEADS + h;
    const float ex = __expf(logit[i] - smax[s]);
    logit[i] = ex;
    atomicAdd(&ssum[s], ex);
}

// ---------------------------------------------------------------------------
// weighted scatter: dst[dI][h][:] += src[sI][h][:] * (ex / (ssum[dI,h]+eps))
// ---------------------------------------------------------------------------
__global__ void seg_scatter_kernel(
    const int* __restrict__ gidx, const int* __restrict__ sidx,
    const float* __restrict__ src, const float* __restrict__ ssum,
    const float* __restrict__ w_ex, float* __restrict__ dst) {

    int i = blockIdx.x * blockDim.x + threadIdx.x;
    if (i >= NNZ_E * HEADS) return;
    const int nz = i >> 3, h = i & 7;
    const int sI = gidx[nz], dI = sidx[nz];
    const float w = w_ex[i] / (ssum[(size_t)dI * HEADS + h] + 1e-16f);

    const float4* s4 = (const float4*)(src + (size_t)sI * HC + h * CH);
    float*       drow = dst + (size_t)dI * HC + h * CH;
#pragma unroll
    for (int q = 0; q < 4; ++q) {
        const float4 x = s4[q];
        atomicAdd(&drow[4 * q + 0], x.x * w);
        atomicAdd(&drow[4 * q + 1], x.y * w);
        atomicAdd(&drow[4 * q + 2], x.z * w);
        atomicAdd(&drow[4 * q + 3], x.w * w);
    }
}

// ---------------------------------------------------------------------------
// final relu in place
// ---------------------------------------------------------------------------
__global__ void relu_kernel(float* __restrict__ p, int n) {
    int i = blockIdx.x * blockDim.x + threadIdx.x;
    if (i < n) p[i] = fmaxf(p[i], 0.0f);
}

// ---------------------------------------------------------------------------
extern "C" void kernel_launch(void* const* d_in, const int* in_sizes, int n_in,
                              void* d_out, int out_size, void* d_ws, size_t ws_size,
                              hipStream_t stream) {
    const float* X     = (const float*)d_in[0];
    const float* Ww    = (const float*)d_in[1];
    const float* Wb    = (const float*)d_in[2];
    const float* att_e = (const float*)d_in[3];
    const float* att_v = (const float*)d_in[4];
    const int* vertex  = (const int*)d_in[5];
    const int* edges   = (const int*)d_in[6];
    const int* e_class = (const int*)d_in[7];
    const int* v_class = (const int*)d_in[8];
    float* out = (float*)d_out;

    // workspace layout (floats): 37.6M floats + 16K f16 = 150.4 MB + 32 KB
    float* ws    = (float*)d_ws;
    float* Xn    = ws;                              // N*HC   = 12.8M
    float* Xe    = Xn    + (size_t)N_NODES * HC;    // E*HC   =  6.4M
    float* beta  = Xe    + (size_t)E_EDGES * HC;    // NNZ*H  =  8.0M
    float* alpha = beta  + (size_t)NNZ_E   * HEADS; // NNZ*H  =  8.0M
    float* emax  = alpha + (size_t)NNZ_E   * HEADS; // E*H    =  0.4M
    float* esum  = emax  + (size_t)E_EDGES * HEADS; // E*H    =  0.4M
    float* vmax  = esum  + (size_t)E_EDGES * HEADS; // N*H    =  0.8M
    float* vsum  = vmax  + (size_t)N_NODES * HEADS; // N*H    =  0.8M
    _Float16* Wh = (_Float16*)(vsum + (size_t)N_NODES * HEADS); // 16384 f16 (32B-aligned)

    const int TB = 256;
    auto nb = [](long n) { return (int)((n + 255) / 256); };

    // ---- init accumulators (rewritten deterministically every call) ----
    fill_kernel<<<nb((long)E_EDGES * HC), TB, 0, stream>>>(Xe, 0.0f, E_EDGES * HC);
    fill_kernel<<<nb((long)N_NODES * HC), TB, 0, stream>>>(out, 0.0f, N_NODES * HC);
    fill_kernel<<<nb((long)E_EDGES * HEADS), TB, 0, stream>>>(emax, -INFINITY, E_EDGES * HEADS);
    fill_kernel<<<nb((long)E_EDGES * HEADS), TB, 0, stream>>>(esum, 0.0f, E_EDGES * HEADS);
    fill_kernel<<<nb((long)N_NODES * HEADS), TB, 0, stream>>>(vmax, -INFINITY, N_NODES * HEADS);
    fill_kernel<<<nb((long)N_NODES * HEADS), TB, 0, stream>>>(vsum, 0.0f, N_NODES * HEADS);

    // ---- pack W into f16 WMMA B-fragment order, then WMMA projection ----
    pack_w_kernel<<<nb((long)HC * IN_CH), TB, 0, stream>>>(Ww, Wh);
    gemm_wmma_kernel<<<N_NODES / 16, 256, 0, stream>>>(X, Wh, Wb, Xn);

    const long NE = (long)NNZ_E * HEADS;

    // ---- vertex -> edge attention + softmax over `edges` segments ----
    seg_logits_kernel<<<nb(NE), TB, 0, stream>>>(Xn, att_e, vertex, edges, e_class, beta, emax);
    seg_exp_kernel<<<nb(NE), TB, 0, stream>>>(edges, emax, beta, esum);
    seg_scatter_kernel<<<nb(NE), TB, 0, stream>>>(vertex, edges, Xn, esum, beta, Xe);

    // ---- edge -> vertex attention + softmax over `vertex` segments ----
    seg_logits_kernel<<<nb(NE), TB, 0, stream>>>(Xe, att_v, edges, vertex, v_class, alpha, vmax);
    seg_exp_kernel<<<nb(NE), TB, 0, stream>>>(vertex, vmax, alpha, vsum);
    seg_scatter_kernel<<<nb(NE), TB, 0, stream>>>(edges, vertex, Xe, vsum, alpha, out);

    // ---- final relu ----
    relu_kernel<<<nb((long)N_NODES * HC), TB, 0, stream>>>(out, N_NODES * HC);
}
